// Darts_8967891714120
// MI455X (gfx1250) — compile-verified, matrix operand order checked by reading, not run on hardware
//
#include <hip/hip_runtime.h>
#include <hip/hip_bf16.h>
#include <cmath>

// ---------------------------------------------------------------------------
// DARTS GNN supernet forward for MI455X (gfx1250, wave32, WMMA).
// Dense GEMMs -> v_wmma_f32_16x16x32_bf16 (bf16 in, f32 acc), 64 cols/block
// (compile-time 4 N-tiles so accumulators stay in fixed VGPRs — no v_movrel).
// Graph aggregations -> CSR (sorted by dst) gather, one 64-thread block/node.
// ---------------------------------------------------------------------------

typedef __attribute__((ext_vector_type(16))) __bf16 v16bf;
typedef __attribute__((ext_vector_type(8)))  float  v8f;

__device__ __forceinline__ unsigned short f2bf(float f) {
    unsigned int u = __float_as_uint(f);
    return (unsigned short)((u + 0x7fffu + ((u >> 16) & 1u)) >> 16);
}
__device__ __forceinline__ float sigmf(float x) { return 1.0f / (1.0f + __expf(-x)); }

// ============================= WMMA GEMM ===================================
// C[M x Nreal] = A[M x K](bf16) @ W (packed as Bt[Npad x K] bf16, Bt[n][k]=W[k][n])
// 8 waves/block; each wave: 16-row strip x 4 fixed N-tiles (64 cols).
// Npad must be a multiple of 64; grid.y = Npad/64.
// A-fragment (16-bit, 16x32): lane<16 -> M=lane, K=8*half.. ; two b128 loads.
union AFrag { v16bf v; uint4 q[2]; };

__global__ __launch_bounds__(256)
void wmma_gemm(const unsigned short* __restrict__ A, const unsigned short* __restrict__ Bt,
               const float* __restrict__ bias, float* __restrict__ C,
               int M, int K, int Nreal, int flags) {
    const int wv   = threadIdx.x >> 5;
    const int lane = threadIdx.x & 31;
    const int hf   = lane >> 4;
    const int lm   = lane & 15;
    const int rbase = blockIdx.x * 128 + wv * 16;
    int r = rbase + lm; if (r >= M) r = M - 1;          // clamp loads; stores guarded
    const int ntile0 = blockIdx.y * 4;

    v8f zero = {0.f,0.f,0.f,0.f,0.f,0.f,0.f,0.f};
    v8f acc0 = zero, acc1 = zero, acc2 = zero, acc3 = zero;

    const unsigned short* arow = A + (size_t)r * K + 8 * hf;
    const unsigned short* b0 = Bt + (size_t)((ntile0 + 0) * 16 + lm) * K + 8 * hf;
    const unsigned short* b1 = Bt + (size_t)((ntile0 + 1) * 16 + lm) * K + 8 * hf;
    const unsigned short* b2 = Bt + (size_t)((ntile0 + 2) * 16 + lm) * K + 8 * hf;
    const unsigned short* b3 = Bt + (size_t)((ntile0 + 3) * 16 + lm) * K + 8 * hf;

    for (int kt = 0; kt < K; kt += 32) {
        AFrag a, b;
        const uint4* pa = (const uint4*)(arow + kt);
        a.q[0] = pa[0];
        a.q[1] = pa[2];                                  // +16 elements (K=16..31)
        const uint4* pb;
        pb = (const uint4*)(b0 + kt); b.q[0] = pb[0]; b.q[1] = pb[2];
        acc0 = __builtin_amdgcn_wmma_f32_16x16x32_bf16(false, a.v, false, b.v, (short)0, acc0, false, false);
        pb = (const uint4*)(b1 + kt); b.q[0] = pb[0]; b.q[1] = pb[2];
        acc1 = __builtin_amdgcn_wmma_f32_16x16x32_bf16(false, a.v, false, b.v, (short)0, acc1, false, false);
        pb = (const uint4*)(b2 + kt); b.q[0] = pb[0]; b.q[1] = pb[2];
        acc2 = __builtin_amdgcn_wmma_f32_16x16x32_bf16(false, a.v, false, b.v, (short)0, acc2, false, false);
        pb = (const uint4*)(b3 + kt); b.q[0] = pb[0]; b.q[1] = pb[2];
        acc3 = __builtin_amdgcn_wmma_f32_16x16x32_bf16(false, a.v, false, b.v, (short)0, acc3, false, false);
    }

    v8f accs[4] = {acc0, acc1, acc2, acc3};
#pragma unroll
    for (int t = 0; t < 4; ++t) {
        int colx = (ntile0 + t) * 16 + lm;
        if (colx >= Nreal) continue;
        float bv = (flags & 1) ? bias[colx] : 0.0f;
#pragma unroll
        for (int vi = 0; vi < 8; ++vi) {
            int row = rbase + vi + 8 * hf;
            if (row >= M) continue;
            size_t off = (size_t)row * Nreal + colx;
            float val = accs[t][vi] + bv;
            if (flags & 2) val += C[off];
            C[off] = val;
        }
    }
}

// Pack W[K][Nreal] (f32 row-major) -> Bt[Ncols][K] (bf16), zero for n>=Nreal.
__global__ void pack_wt(const float* __restrict__ W, unsigned short* __restrict__ Bt,
                        int K, int Nreal, int Ncols) {
    size_t idx = (size_t)blockIdx.x * blockDim.x + threadIdx.x;
    size_t total = (size_t)Ncols * K;
    if (idx >= total) return;
    int nc = (int)(idx / K), k = (int)(idx % K);
    float w = (nc < Nreal) ? W[(size_t)k * Nreal + nc] : 0.0f;
    Bt[idx] = f2bf(w);
}

__global__ void cvt_f2bf(const float* __restrict__ in, unsigned short* __restrict__ out, size_t n) {
    size_t i = (size_t)blockIdx.x * blockDim.x + threadIdx.x;
    if (i < n) out[i] = f2bf(in[i]);
}

// ============================= CSR build ===================================
__global__ void count_deg(const int* __restrict__ dst, int E, int* __restrict__ deg) {
    int i = blockIdx.x * blockDim.x + threadIdx.x;
    if (i < E) atomicAdd(&deg[dst[i]], 1);
}
__global__ void scan_rp(const int* __restrict__ deg, int* __restrict__ rp, int n) {
    __shared__ int sh[1024];
    __shared__ int carry;
    if (threadIdx.x == 0) { carry = 0; rp[0] = 0; }
    __syncthreads();
    for (int base = 0; base < n; base += 1024) {
        int i = base + threadIdx.x;
        int x = (i < n) ? deg[i] : 0;
        sh[threadIdx.x] = x; __syncthreads();
        for (int o = 1; o < 1024; o <<= 1) {
            int t = (threadIdx.x >= o) ? sh[threadIdx.x - o] : 0;
            __syncthreads();
            sh[threadIdx.x] += t;
            __syncthreads();
        }
        if (i < n) rp[i + 1] = carry + sh[threadIdx.x];
        __syncthreads();
        if (threadIdx.x == 1023) carry += sh[1023];
        __syncthreads();
    }
}
__global__ void fill_csr(const int* __restrict__ src, const int* __restrict__ dst, int E,
                         const int* __restrict__ rp, int* __restrict__ fill, int* __restrict__ col) {
    int i = blockIdx.x * blockDim.x + threadIdx.x;
    if (i >= E) return;
    int v = dst[i];
    int pos = rp[v] + atomicAdd(&fill[v], 1);
    col[pos] = src[i];
}
__global__ void deg_floats(const int* __restrict__ deg, float* degf, float* degclip,
                           float* dinv, float* dinvl, float* dvc, int n) {
    int v = blockIdx.x * blockDim.x + threadIdx.x;
    if (v >= n) return;
    float d = (float)deg[v];
    degf[v] = d;
    float dc = d < 1.f ? 1.f : d;
    degclip[v] = dc;
    dinv[v]  = d > 0.f ? rsqrtf(d) : 0.f;
    dinvl[v] = rsqrtf(d + 1.f);
    dvc[v]   = rsqrtf(dc);
}

// ========================= generic CSR aggregation ==========================
// mode: 0=SUM, 1=MEAN(clip deg>=1), 2=GCN w/ self-loop (dinvl), 3=GCN no loop (dinv)
__global__ __launch_bounds__(64)
void spmm_csr(const float* __restrict__ X, float* __restrict__ out,
              const int* __restrict__ rp, const int* __restrict__ col,
              const float* __restrict__ degf, const float* __restrict__ dnv,
              int n, int mode, float oscale, const float* __restrict__ addbuf, float aadd,
              int accum) {
    int v = blockIdx.x, c = threadIdx.x;
    int s = rp[v], e = rp[v + 1];
    float acc = 0.f;
    if (mode == 2) {
        float wv = dnv[v];
        for (int p = s; p < e; ++p) { int u = col[p]; acc += dnv[u] * wv * X[(size_t)u * 64 + c]; }
        acc += wv * wv * X[(size_t)v * 64 + c];
    } else if (mode == 3) {
        float wv = dnv[v];
        for (int p = s; p < e; ++p) { int u = col[p]; acc += dnv[u] * wv * X[(size_t)u * 64 + c]; }
    } else {
        for (int p = s; p < e; ++p) acc += X[(size_t)col[p] * 64 + c];
        if (mode == 1) { float d = degf[v]; acc /= (d > 1.f ? d : 1.f); }
    }
    float r = oscale * acc;
    if (addbuf) r += aadd * addbuf[(size_t)v * 64 + c];
    size_t o = (size_t)v * 64 + c;
    out[o] = accum ? out[o] + r : r;
}

// ===================== softmax aggregation (GAT / AGNN) =====================
__global__ __launch_bounds__(64)
void smax_agg(const float* __restrict__ H, const float* __restrict__ elog,
              const float* __restrict__ selflog, const int* __restrict__ rp,
              const int* __restrict__ col, const float* __restrict__ bias,
              float* __restrict__ out, int n) {
    __shared__ float red[64];
    int v = blockIdx.x, t = threadIdx.x;
    int s = rp[v], e = rp[v + 1];
    float m = -1e30f;
    for (int p = s + t; p < e; p += 64) m = fmaxf(m, elog[p]);
    red[t] = m; __syncthreads();
    for (int o = 32; o > 0; o >>= 1) { if (t < o) red[t] = fmaxf(red[t], red[t + o]); __syncthreads(); }
    m = fmaxf(red[0], selflog[v]); __syncthreads();
    float sum = 0.f;
    for (int p = s + t; p < e; p += 64) sum += __expf(elog[p] - m);
    red[t] = sum; __syncthreads();
    for (int o = 32; o > 0; o >>= 1) { if (t < o) red[t] += red[t + o]; __syncthreads(); }
    float wself = __expf(selflog[v] - m);
    float inv = 1.0f / (red[0] + wself + 1e-16f);
    float acc = 0.f;
    for (int p = s; p < e; ++p)
        acc += __expf(elog[p] - m) * inv * H[(size_t)col[p] * 64 + t];
    acc += wself * inv * H[(size_t)v * 64 + t];
    if (bias) acc += bias[t];
    out[(size_t)v * 64 + t] = acc;
}

__global__ __launch_bounds__(64)
void gat_logits(const float* __restrict__ s, const float* __restrict__ d,
                const int* __restrict__ rp, const int* __restrict__ col,
                float* __restrict__ elog, float* __restrict__ selflog, int n) {
    int v = blockIdx.x, t = threadIdx.x;
    int st = rp[v], e = rp[v + 1];
    float dv = d[v];
    for (int p = st + t; p < e; p += 64) {
        float x = s[col[p]] + dv;
        elog[p] = x > 0.f ? x : 0.2f * x;
    }
    if (t == 0) { float x = s[v] + dv; selflog[v] = x > 0.f ? x : 0.2f * x; }
}

__global__ void agnn_norm(const float* __restrict__ H, const float* __restrict__ beta,
                          float* __restrict__ ninv, float* __restrict__ selflog, int n) {
    int v = blockIdx.x * blockDim.x + threadIdx.x;
    if (v >= n) return;
    float s = 0.f;
    for (int c = 0; c < 64; ++c) { float h = H[(size_t)v * 64 + c]; s += h * h; }
    float iv = 1.0f / (sqrtf(s) + 1e-16f);
    ninv[v] = iv;
    selflog[v] = beta[0] * (s * iv * iv);
}
__global__ __launch_bounds__(64)
void agnn_logits(const float* __restrict__ H, const float* __restrict__ ninv,
                 const float* __restrict__ beta, const int* __restrict__ rp,
                 const int* __restrict__ col, float* __restrict__ elog, int n) {
    __shared__ float red[64];
    int v = blockIdx.x, t = threadIdx.x;
    int s = rp[v], e = rp[v + 1];
    float hv = H[(size_t)v * 64 + t];
    float b = beta[0], nv = ninv[v];
    for (int p = s; p < e; ++p) {
        int u = col[p];
        red[t] = hv * H[(size_t)u * 64 + t];
        __syncthreads();
        for (int o = 32; o > 0; o >>= 1) { if (t < o) red[t] += red[t + o]; __syncthreads(); }
        if (t == 0) elog[p] = b * red[0] * nv * ninv[u];
        __syncthreads();
    }
}

// =============================== FeaSt ======================================
__global__ __launch_bounds__(64)
void feast_agg(const float* __restrict__ H2 /*[n][128]*/, const float* __restrict__ pu,
               const float* __restrict__ cpar, const float* __restrict__ bias,
               const float* __restrict__ degf, const int* __restrict__ rp,
               const int* __restrict__ col, float* __restrict__ out, int n) {
    int v = blockIdx.x, t = threadIdx.x;
    float c0 = cpar[0], c1 = cpar[1];
    float pv0 = pu[2 * v], pv1 = pu[2 * v + 1];
    int s = rp[v], e = rp[v + 1];
    float acc = 0.f;
    for (int p = s; p < e; ++p) {
        int u = col[p];
        float t0 = pu[2 * u] - pv0 + c0, t1 = pu[2 * u + 1] - pv1 + c1;
        float m = fmaxf(t0, t1);
        float e0 = __expf(t0 - m), e1 = __expf(t1 - m);
        float inv = 1.0f / (e0 + e1);
        acc += (e0 * inv) * H2[(size_t)u * 128 + t] + (e1 * inv) * H2[(size_t)u * 128 + 64 + t];
    }
    { // self edge: src==dst -> t = c
        float m = fmaxf(c0, c1);
        float e0 = __expf(c0 - m), e1 = __expf(c1 - m);
        float inv = 1.0f / (e0 + e1);
        acc += (e0 * inv) * H2[(size_t)v * 128 + t] + (e1 * inv) * H2[(size_t)v * 128 + 64 + t];
    }
    out[(size_t)v * 64 + t] = acc / (degf[v] + 1.0f) + bias[t];
}

// =============================== GENConv ====================================
__global__ __launch_bounds__(64)
void gen_agg(const float* __restrict__ X, const int* __restrict__ rp,
             const int* __restrict__ col, float* __restrict__ out, int n) {
    int v = blockIdx.x, t = threadIdx.x;
    int s = rp[v], e = rp[v + 1];
    float m = -1e30f;
    for (int p = s; p < e; ++p) {
        float x = X[(size_t)col[p] * 64 + t];
        x = (x > 0.f ? x : 0.f) + 1e-7f;
        m = fmaxf(m, x);
    }
    float sum = 0.f;
    for (int p = s; p < e; ++p) {
        float x = X[(size_t)col[p] * 64 + t];
        x = (x > 0.f ? x : 0.f) + 1e-7f;
        sum += __expf(x - m);
    }
    sum += 1e-16f;
    float acc = 0.f;
    for (int p = s; p < e; ++p) {
        float x = X[(size_t)col[p] * 64 + t];
        x = (x > 0.f ? x : 0.f) + 1e-7f;
        acc += (__expf(x - m) / sum) * x;
    }
    out[(size_t)v * 64 + t] = acc;
}

// ================================ GMM =======================================
__global__ __launch_bounds__(64)
void gmm_agg(const float* __restrict__ HG /*[n][256]*/, const float* __restrict__ dvc,
             const float* __restrict__ mu, const float* __restrict__ sigma,
             const float* __restrict__ degclip, const int* __restrict__ rp,
             const int* __restrict__ col, float* __restrict__ out, int n) {
    __shared__ float smu[8], ssg[8];
    int v = blockIdx.x, t = threadIdx.x;
    if (t < 8) { smu[t] = mu[t]; ssg[t] = sigma[t]; }
    __syncthreads();
    float dvv = dvc[v];
    int s = rp[v], e = rp[v + 1];
    float acc = 0.f;
    for (int p = s; p < e; ++p) {
        int u = col[p];
        float du = dvc[u];
        for (int k = 0; k < 4; ++k) {
            float a0 = (du - smu[2 * k]) / ssg[2 * k];
            float a1 = (dvv - smu[2 * k + 1]) / ssg[2 * k + 1];
            float g = __expf(-0.5f * (a0 * a0 + a1 * a1));
            acc += g * HG[(size_t)u * 256 + k * 64 + t];
        }
    }
    out[(size_t)v * 64 + t] += acc / degclip[v];
}

// ============================== GIN BN ======================================
__global__ __launch_bounds__(256)
void colstats(const float* __restrict__ H, float* __restrict__ stats, int n) {
    __shared__ float sm[256], sq[256];
    int c = threadIdx.x & 63, g = threadIdx.x >> 6;
    float s = 0.f, q = 0.f;
    for (int v = blockIdx.x * 4 + g; v < n; v += gridDim.x * 4) {
        float x = H[(size_t)v * 64 + c];
        s += x; q += x * x;
    }
    sm[threadIdx.x] = s; sq[threadIdx.x] = q; __syncthreads();
    if (g == 0) {
        s = sm[c] + sm[64 + c] + sm[128 + c] + sm[192 + c];
        q = sq[c] + sq[64 + c] + sq[128 + c] + sq[192 + c];
        atomicAdd(&stats[c], s);
        atomicAdd(&stats[64 + c], q);
    }
}
__global__ void bn_apply(const float* __restrict__ H, const float* __restrict__ stats,
                         const float* __restrict__ g, const float* __restrict__ b,
                         unsigned short* __restrict__ out, int n) {
    size_t idx = (size_t)blockIdx.x * blockDim.x + threadIdx.x;
    size_t tot = (size_t)n * 64;
    if (idx >= tot) return;
    int c = (int)(idx & 63);
    float invn = 1.0f / (float)n;
    float mu = stats[c] * invn;
    float var = stats[64 + c] * invn - mu * mu;
    float x = (H[idx] - mu) * rsqrtf(var + 1e-5f) * g[c] + b[c];
    out[idx] = f2bf(x > 0.f ? x : 0.f);
}

// ============================ elementwise ===================================
__global__ void axpy_ptr(float* __restrict__ y, const float* __restrict__ x,
                         const float* __restrict__ a, size_t n) {
    size_t i = (size_t)blockIdx.x * blockDim.x + threadIdx.x;
    if (i < n) y[i] += a[0] * x[i];
}
__global__ void lincomb2(float* __restrict__ o, const float* __restrict__ x, float a,
                         const float* __restrict__ y, float b, size_t n) {
    size_t i = (size_t)blockIdx.x * blockDim.x + threadIdx.x;
    if (i < n) o[i] = a * x[i] + b * y[i];
}
__global__ void gin_comb(float* __restrict__ o, const float* __restrict__ x,
                         const float* __restrict__ agg, const float* __restrict__ eps, size_t n) {
    size_t i = (size_t)blockIdx.x * blockDim.x + threadIdx.x;
    if (i < n) o[i] = (1.0f + eps[0]) * x[i] + agg[i];
}
__global__ void relu_ip(float* __restrict__ x, size_t n) {
    size_t i = (size_t)blockIdx.x * blockDim.x + threadIdx.x;
    if (i < n) x[i] = x[i] > 0.f ? x[i] : 0.f;
}
__global__ void add_bias64(float* __restrict__ y, const float* __restrict__ b, size_t n) {
    size_t i = (size_t)blockIdx.x * blockDim.x + threadIdx.x;
    if (i < n) y[i] += b[i & 63];
}
__global__ void copy_f32(float* __restrict__ d, const float* __restrict__ s, size_t n) {
    size_t i = (size_t)blockIdx.x * blockDim.x + threadIdx.x;
    if (i < n) d[i] = s[i];
}
__global__ void sum6(float* __restrict__ o, const float* a, const float* b, const float* c,
                     const float* d, const float* e, const float* f, size_t n) {
    size_t i = (size_t)blockIdx.x * blockDim.x + threadIdx.x;
    if (i < n) o[i] = a[i] + b[i] + c[i] + d[i] + e[i] + f[i];
}
__global__ void gated_gru(const float* __restrict__ gi, const float* __restrict__ gh,
                          float* __restrict__ h, size_t n64) {
    size_t idx = (size_t)blockIdx.x * blockDim.x + threadIdx.x;
    if (idx >= n64) return;
    size_t v = idx >> 6; int c = (int)(idx & 63);
    const float* GI = gi + v * 192; const float* GH = gh + v * 192;
    float r = sigmf(GI[c] + GH[c]);
    float z = sigmf(GI[64 + c] + GH[64 + c]);
    float g = tanhf(GI[128 + c] + r * GH[128 + c]);
    h[idx] = (1.f - z) * g + z * h[idx];
}
__global__ void gemv_small(const float* __restrict__ X, const float* __restrict__ U,
                           const float* __restrict__ c0, float* __restrict__ out, int n, int R) {
    int v = blockIdx.x * blockDim.x + threadIdx.x;
    if (v >= n) return;
    for (int r = 0; r < R; ++r) {
        float s = c0 ? c0[r] : 0.f;
        for (int c = 0; c < 64; ++c) s += X[(size_t)v * 64 + c] * U[c * R + r];
        out[(size_t)v * R + r] = s;
    }
}

// ========================== mixture kernels =================================
__global__ void compute_weights(const float* alpha, const float* gamma,
                                const float* b1, const float* b2, const float* b3,
                                const float* b4, const float* b5, const float* b6,
                                float* wbuf) {
    if (blockIdx.x || threadIdx.x) return;
    const float* bs[6] = {b1, b2, b3, b4, b5, b6};
    {
        float m = -1e30f; for (int i = 0; i < 5; ++i) m = fmaxf(m, alpha[i]);
        float s = 0.f, e[5];
        for (int i = 0; i < 5; ++i) { e[i] = __expf(alpha[i] - m); s += e[i]; }
        for (int i = 0; i < 5; ++i) wbuf[i] = e[i] / s;
    }
    {
        float m = -1e30f; for (int i = 0; i < 5; ++i) m = fmaxf(m, gamma[i]);
        float s = 0.f, e[5];
        for (int i = 0; i < 5; ++i) { e[i] = __expf(gamma[i] - m); s += e[i]; }
        for (int i = 0; i < 5; ++i) wbuf[5 + i] = e[i] / s;
    }
    int pair = 0;
    for (int j = 1; j <= 6; ++j)
        for (int i = 0; i < j; ++i) {
            const float* sp = bs[j - 1] + i * 12 + 1;
            float m = -1e30f; for (int k = 0; k < 12; ++k) m = fmaxf(m, sp[k]);
            float s = 0.f, e[12];
            for (int k = 0; k < 12; ++k) { e[k] = __expf(sp[k] - m); s += e[k]; }
            for (int k = 0; k < 12; ++k) wbuf[10 + pair * 12 + k] = e[k] / s;
            pair++;
        }
}

__global__ __launch_bounds__(64)
void mix5(const float* __restrict__ Hcat, const float* __restrict__ a, float* __restrict__ out, int n) {
    __shared__ float red[64];
    int v = blockIdx.x, t = threadIdx.x;
    const float* row = Hcat + (size_t)v * 320;
    float x0 = row[t], x1 = row[64 + t], x2 = row[128 + t], x3 = row[192 + t], x4 = row[256 + t];
    red[t] = x3; __syncthreads();
    for (int o = 32; o > 0; o >>= 1) { if (t < o) red[t] = fmaxf(red[t], red[t + o]); __syncthreads(); }
    float m = red[0]; __syncthreads();
    float ex = __expf(x3 - m);
    red[t] = ex; __syncthreads();
    for (int o = 32; o > 0; o >>= 1) { if (t < o) red[t] += red[t + o]; __syncthreads(); }
    float sm = ex / red[0];
    out[(size_t)v * 64 + t] = a[0] * sigmf(x0) + a[1] * tanhf(x1) +
                              a[2] * fmaxf(x2, 0.f) + a[3] * sm + a[4] * x4;
}

__global__ __launch_bounds__(64)
void final_mix(const float* __restrict__ Z, const float* __restrict__ g, float* __restrict__ out, int n) {
    __shared__ float red[64];
    int v = blockIdx.x, t = threadIdx.x;
    const float* row = Z + (size_t)v * 200;
    float x0 = 0, x1 = 0, x2 = 0, x3 = -1e30f, x4 = 0;
    if (t < 40) { x0 = row[t]; x1 = row[40 + t]; x2 = row[80 + t]; x3 = row[120 + t]; x4 = row[160 + t]; }
    red[t] = x3; __syncthreads();
    for (int o = 32; o > 0; o >>= 1) { if (t < o) red[t] = fmaxf(red[t], red[t + o]); __syncthreads(); }
    float m = red[0]; __syncthreads();
    float ex = (t < 40) ? __expf(x3 - m) : 0.f;
    red[t] = ex; __syncthreads();
    for (int o = 32; o > 0; o >>= 1) { if (t < o) red[t] += red[t + o]; __syncthreads(); }
    if (t < 40) {
        float sm = ex / red[0];
        out[(size_t)v * 40 + t] = g[0] * sigmf(x0) + g[1] * tanhf(x1) +
                                  g[2] * fmaxf(x2, 0.f) + g[3] * sm + g[4] * x4;
    }
}

// ============================== host side ===================================
static inline unsigned blk(size_t total) { return (unsigned)((total + 255) / 256); }

struct Ctx {
    hipStream_t st;
    int n, E;
    int *deg, *rp, *fill, *col;
    float *degf, *degclip, *dinv, *dinvl, *dvc;
    unsigned short *Xb, *bfA, *bfB, *wpack;
    float *bigA, *bigB, *T, *H, *S, *A64, *z0, *z1, *hcur;
    float *elog, *selflog, *svec, *dvec, *pu, *ninv, *stats, *bias320, *bias200, *wbuf;
    float *ys[7];
    unsigned short *ysb[7];
};

static void gemm_h(const Ctx& c, const unsigned short* A, const float* W, const float* bias,
                   float* C, int K, int Nreal, bool accum) {
    int Npad = (Nreal + 63) & ~63;   // kernel computes 64 cols/block (4 fixed tiles)
    pack_wt<<<blk((size_t)Npad * K), 256, 0, c.st>>>(W, c.wpack, K, Nreal, Npad);
    dim3 grid((c.n + 127) / 128, Npad / 64);
    int flags = (bias ? 1 : 0) | (accum ? 2 : 0);
    wmma_gemm<<<grid, 256, 0, c.st>>>(A, c.wpack, bias, C, c.n, K, Nreal, flags);
}

static void run_pair(const Ctx& c, void* const* P, const float* X, const unsigned short* Xb,
                     float* ysj, const float* bw) {
    const int n = c.n;
    const size_t n64 = (size_t)n * 64;
    hipStream_t st = c.st;

    // --- t=0 GAT: W,a_src,a_dst,b ---
    gemm_h(c, Xb, (const float*)P[0], nullptr, c.H, 64, 64, false);
    gemv_small<<<blk(n), 256, 0, st>>>(c.H, (const float*)P[1], nullptr, c.svec, n, 1);
    gemv_small<<<blk(n), 256, 0, st>>>(c.H, (const float*)P[2], nullptr, c.dvec, n, 1);
    gat_logits<<<n, 64, 0, st>>>(c.svec, c.dvec, c.rp, c.col, c.elog, c.selflog, n);
    smax_agg<<<n, 64, 0, st>>>(c.H, c.elog, c.selflog, c.rp, c.col, (const float*)P[3], c.T, n);
    axpy_ptr<<<blk(n64), 256, 0, st>>>(ysj, c.T, bw + 0, n64);

    // --- t=1 GIN: eps,W1,b1,bn_g,bn_b,W2,b2 ---
    spmm_csr<<<n, 64, 0, st>>>(X, c.A64, c.rp, c.col, c.degf, c.dinv, n, 0, 1.f, nullptr, 0.f, 0);
    gin_comb<<<blk(n64), 256, 0, st>>>(c.T, X, c.A64, (const float*)P[4], n64);
    cvt_f2bf<<<blk(n64), 256, 0, st>>>(c.T, c.bfA, n64);
    gemm_h(c, c.bfA, (const float*)P[5], (const float*)P[6], c.H, 64, 64, false);
    hipMemsetAsync(c.stats, 0, 128 * sizeof(float), st);
    colstats<<<256, 256, 0, st>>>(c.H, c.stats, n);
    bn_apply<<<blk(n64), 256, 0, st>>>(c.H, c.stats, (const float*)P[7], (const float*)P[8], c.bfA, n);
    gemm_h(c, c.bfA, (const float*)P[9], (const float*)P[10], c.T, 64, 64, false);
    axpy_ptr<<<blk(n64), 256, 0, st>>>(ysj, c.T, bw + 1, n64);

    // --- t=2 SAGE: Wl,bl,Wr ---
    spmm_csr<<<n, 64, 0, st>>>(X, c.A64, c.rp, c.col, c.degf, c.dinv, n, 1, 1.f, nullptr, 0.f, 0);
    cvt_f2bf<<<blk(n64), 256, 0, st>>>(c.A64, c.bfA, n64);
    gemm_h(c, c.bfA, (const float*)P[11], (const float*)P[12], c.T, 64, 64, false);
    gemm_h(c, Xb, (const float*)P[13], nullptr, c.T, 64, 64, true);
    axpy_ptr<<<blk(n64), 256, 0, st>>>(ysj, c.T, bw + 2, n64);

    // --- t=3 GCN: W,b ---
    gemm_h(c, Xb, (const float*)P[14], nullptr, c.H, 64, 64, false);
    spmm_csr<<<n, 64, 0, st>>>(c.H, c.T, c.rp, c.col, c.degf, c.dinvl, n, 2, 1.f, nullptr, 0.f, 0);
    add_bias64<<<blk(n64), 256, 0, st>>>(c.T, (const float*)P[15], n64);
    axpy_ptr<<<blk(n64), 256, 0, st>>>(ysj, c.T, bw + 3, n64);

    // --- t=4 SGC: W,b ---
    spmm_csr<<<n, 64, 0, st>>>(X, c.S, c.rp, c.col, c.degf, c.dinvl, n, 2, 1.f, nullptr, 0.f, 0);
    cvt_f2bf<<<blk(n64), 256, 0, st>>>(c.S, c.bfA, n64);
    gemm_h(c, c.bfA, (const float*)P[16], (const float*)P[17], c.T, 64, 64, false);
    axpy_ptr<<<blk(n64), 256, 0, st>>>(ysj, c.T, bw + 4, n64);

    // --- t=5 APPNP: W,b ---
    gemm_h(c, Xb, (const float*)P[18], (const float*)P[19], c.H, 64, 64, false);
    copy_f32<<<blk(n64), 256, 0, st>>>(c.z0, c.H, n64);
    {
        float* zin = c.z0; float* zout = c.z1;
        for (int it = 0; it < 10; ++it) {
            spmm_csr<<<n, 64, 0, st>>>(zin, zout, c.rp, c.col, c.degf, c.dinvl, n, 2,
                                       0.9f, c.H, 0.1f, 0);
            float* tmp = zin; zin = zout; zout = tmp;
        }
        axpy_ptr<<<blk(n64), 256, 0, st>>>(ysj, zin, bw + 5, n64);
    }

    // --- t=6 AGNN: W,b,beta ---
    gemm_h(c, Xb, (const float*)P[20], (const float*)P[21], c.H, 64, 64, false);
    agnn_norm<<<blk(n), 256, 0, st>>>(c.H, (const float*)P[22], c.ninv, c.selflog, n);
    agnn_logits<<<n, 64, 0, st>>>(c.H, c.ninv, (const float*)P[22], c.rp, c.col, c.elog, n);
    smax_agg<<<n, 64, 0, st>>>(c.H, c.elog, c.selflog, c.rp, c.col, nullptr, c.T, n);
    axpy_ptr<<<blk(n64), 256, 0, st>>>(ysj, c.T, bw + 6, n64);

    // --- t=7 ARMA: W,V,b ---
    spmm_csr<<<n, 64, 0, st>>>(X, c.S, c.rp, c.col, c.degf, c.dinv, n, 3, 1.f, nullptr, 0.f, 0);
    cvt_f2bf<<<blk(n64), 256, 0, st>>>(c.S, c.bfA, n64);
    gemm_h(c, c.bfA, (const float*)P[23], (const float*)P[25], c.T, 64, 64, false);
    gemm_h(c, Xb, (const float*)P[24], nullptr, c.T, 64, 64, true);
    relu_ip<<<blk(n64), 256, 0, st>>>(c.T, n64);
    axpy_ptr<<<blk(n64), 256, 0, st>>>(ysj, c.T, bw + 7, n64);

    // --- t=8 FEAST: u,c,W,b ---
    pack_wt<<<blk(64 * 64), 256, 0, st>>>((const float*)P[28], c.wpack, 64, 64, 64);
    pack_wt<<<blk(64 * 64), 256, 0, st>>>((const float*)P[28] + 4096, c.wpack + 64 * 64, 64, 64, 64);
    {
        dim3 grid((n + 127) / 128, 2);
        wmma_gemm<<<grid, 256, 0, st>>>(Xb, c.wpack, nullptr, c.bigB, n, 64, 128, 0);
    }
    gemv_small<<<blk(n), 256, 0, st>>>(X, (const float*)P[26], nullptr, c.pu, n, 2);
    feast_agg<<<n, 64, 0, st>>>(c.bigB, c.pu, (const float*)P[27], (const float*)P[29],
                                c.degf, c.rp, c.col, c.T, n);
    axpy_ptr<<<blk(n64), 256, 0, st>>>(ysj, c.T, bw + 8, n64);

    // --- t=9 GEN: W1,b1,W2,b2 ---
    gen_agg<<<n, 64, 0, st>>>(X, c.rp, c.col, c.A64, n);
    lincomb2<<<blk(n64), 256, 0, st>>>(c.T, X, 1.f, c.A64, 1.f, n64);
    cvt_f2bf<<<blk(n64), 256, 0, st>>>(c.T, c.bfA, n64);
    gemm_h(c, c.bfA, (const float*)P[30], (const float*)P[31], c.bigB, 64, 128, false);
    relu_ip<<<blk((size_t)n * 128), 256, 0, st>>>(c.bigB, (size_t)n * 128);
    cvt_f2bf<<<blk((size_t)n * 128), 256, 0, st>>>(c.bigB, c.bfB, (size_t)n * 128);
    gemm_h(c, c.bfB, (const float*)P[32], (const float*)P[33], c.T, 128, 64, false);
    axpy_ptr<<<blk(n64), 256, 0, st>>>(ysj, c.T, bw + 9, n64);

    // --- t=10 GMM: Wg,mu,sigma,Wroot,b ---
    gemm_h(c, Xb, (const float*)P[34], nullptr, c.bigA, 64, 256, false);
    gemm_h(c, Xb, (const float*)P[37], (const float*)P[38], c.T, 64, 64, false);
    gmm_agg<<<n, 64, 0, st>>>(c.bigA, c.dvc, (const float*)P[35], (const float*)P[36],
                              c.degclip, c.rp, c.col, c.T, n);
    axpy_ptr<<<blk(n64), 256, 0, st>>>(ysj, c.T, bw + 10, n64);

    // --- t=11 GATED: Wl,Wih,bih,Whh,bhh ---
    copy_f32<<<blk(n64), 256, 0, st>>>(c.hcur, X, n64);
    for (int l = 0; l < 3; ++l) {
        cvt_f2bf<<<blk(n64), 256, 0, st>>>(c.hcur, c.bfA, n64);
        gemm_h(c, c.bfA, (const float*)P[39] + (size_t)l * 4096, nullptr, c.A64, 64, 64, false);
        spmm_csr<<<n, 64, 0, st>>>(c.A64, c.S, c.rp, c.col, c.degf, c.dinv, n, 0, 1.f, nullptr, 0.f, 0);
        cvt_f2bf<<<blk(n64), 256, 0, st>>>(c.S, c.bfB, n64);
        gemm_h(c, c.bfB, (const float*)P[40], (const float*)P[41], c.bigA, 64, 192, false);
        gemm_h(c, c.bfA, (const float*)P[42], (const float*)P[43], c.bigB, 64, 192, false);
        gated_gru<<<blk(n64), 256, 0, st>>>(c.bigA, c.bigB, c.hcur, n64);
    }
    axpy_ptr<<<blk(n64), 256, 0, st>>>(ysj, c.hcur, bw + 11, n64);
}

extern "C" void kernel_launch(void* const* d_in, const int* in_sizes, int n_in,
                              void* d_out, int out_size, void* d_ws, size_t ws_size,
                              hipStream_t stream) {
    const int n = in_sizes[0] / 512;
    const int E = in_sizes[1] / 2;
    const size_t n64 = (size_t)n * 64;

    const float* x = (const float*)d_in[0];
    const int* src = (const int*)d_in[1];
    const int* dst = src + E;
    const float* alpha = (const float*)d_in[2];
    const float* gamma = (const float*)d_in[3];
    const float* betas[6];
    for (int j = 0; j < 6; ++j) betas[j] = (const float*)d_in[4 + j];
    // xW at [10..19] (W,b)*5 ; zW at [20..29]; convs start at 30, 44 tensors/pair.

    // ---- workspace bump allocator (256B aligned) ----
    char* base = (char*)d_ws;
    size_t off = 0;
    auto alloc = [&](size_t bytes) -> void* {
        size_t a = (off + 255) & ~(size_t)255;
        void* p = base + a;
        off = a + bytes;
        return p;
    };

    Ctx c;
    c.st = stream; c.n = n; c.E = E;
    c.wbuf    = (float*)alloc(262 * 4);
    c.deg     = (int*)alloc((size_t)n * 4);
    c.rp      = (int*)alloc((size_t)(n + 1) * 4);
    c.fill    = (int*)alloc((size_t)n * 4);
    c.col     = (int*)alloc((size_t)E * 4);
    c.degf    = (float*)alloc((size_t)n * 4);
    c.degclip = (float*)alloc((size_t)n * 4);
    c.dinv    = (float*)alloc((size_t)n * 4);
    c.dinvl   = (float*)alloc((size_t)n * 4);
    c.dvc     = (float*)alloc((size_t)n * 4);
    c.Xb      = (unsigned short*)alloc((size_t)n * 512 * 2);
    c.bfA     = (unsigned short*)alloc((size_t)n * 256 * 2);
    c.bfB     = (unsigned short*)alloc((size_t)n * 256 * 2);
    c.wpack   = (unsigned short*)alloc((size_t)320 * 512 * 2);
    c.bigA    = (float*)alloc((size_t)n * 320 * 4);
    c.bigB    = (float*)alloc((size_t)n * 256 * 4);
    c.T       = (float*)alloc(n64 * 4);
    c.H       = (float*)alloc(n64 * 4);
    c.S       = (float*)alloc(n64 * 4);
    c.A64     = (float*)alloc(n64 * 4);
    c.z0      = (float*)alloc(n64 * 4);
    c.z1      = (float*)alloc(n64 * 4);
    c.hcur    = (float*)alloc(n64 * 4);
    c.elog    = (float*)alloc((size_t)E * 4);
    c.selflog = (float*)alloc((size_t)n * 4);
    c.svec    = (float*)alloc((size_t)n * 4);
    c.dvec    = (float*)alloc((size_t)n * 4);
    c.pu      = (float*)alloc((size_t)n * 2 * 4);
    c.ninv    = (float*)alloc((size_t)n * 4);
    c.stats   = (float*)alloc(128 * 4);
    c.bias320 = (float*)alloc(320 * 4);
    c.bias200 = (float*)alloc(200 * 4);
    for (int i = 0; i < 7; ++i) c.ys[i]  = (float*)alloc(n64 * 4);
    for (int i = 0; i < 7; ++i) c.ysb[i] = (unsigned short*)alloc(n64 * 2);

    // ---- mixture weights ----
    compute_weights<<<1, 1, 0, stream>>>(alpha, gamma, betas[0], betas[1], betas[2],
                                         betas[3], betas[4], betas[5], c.wbuf);

    // ---- CSR build (sorted by dst) ----
    hipMemsetAsync(c.deg, 0, (size_t)n * 4, stream);
    count_deg<<<blk(E), 256, 0, stream>>>(dst, E, c.deg);
    scan_rp<<<1, 1024, 0, stream>>>(c.deg, c.rp, n);
    hipMemsetAsync(c.fill, 0, (size_t)n * 4, stream);
    fill_csr<<<blk(E), 256, 0, stream>>>(src, dst, E, c.rp, c.fill, c.col);
    deg_floats<<<blk(n), 256, 0, stream>>>(c.deg, c.degf, c.degclip, c.dinv, c.dinvl, c.dvc, n);

    // ---- input features -> bf16 ----
    cvt_f2bf<<<blk((size_t)n * 512), 256, 0, stream>>>(x, c.Xb, (size_t)n * 512);

    // ---- h0 = sum_i a_i * ACT_i(x @ xW_i + b_i): one 512->320 WMMA GEMM ----
    for (int i = 0; i < 5; ++i) {
        pack_wt<<<blk((size_t)64 * 512), 256, 0, stream>>>(
            (const float*)d_in[10 + 2 * i], c.wpack + (size_t)i * 64 * 512, 512, 64, 64);
        copy_f32<<<1, 64, 0, stream>>>(c.bias320 + i * 64, (const float*)d_in[11 + 2 * i], 64);
    }
    {
        dim3 grid((n + 127) / 128, 5);
        wmma_gemm<<<grid, 256, 0, stream>>>(c.Xb, c.wpack, c.bias320, c.bigA, n, 512, 320, 1);
    }
    mix5<<<n, 64, 0, stream>>>(c.bigA, c.wbuf, c.ys[0], n);
    cvt_f2bf<<<blk(n64), 256, 0, stream>>>(c.ys[0], c.ysb[0], n64);

    // ---- supernet cell edges ----
    int pair = 0;
    for (int j = 1; j <= 6; ++j) {
        hipMemsetAsync(c.ys[j], 0, n64 * 4, stream);
        for (int i = 0; i < j; ++i) {
            run_pair(c, d_in + 30 + (size_t)pair * 44, c.ys[i], c.ysb[i], c.ys[j],
                     c.wbuf + 10 + pair * 12);
            pair++;
        }
        cvt_f2bf<<<blk(n64), 256, 0, stream>>>(c.ys[j], c.ysb[j], n64);
    }

    // ---- readout: h = sum ys[1..6]; out = sum_i g_i ACT_i(h @ zW_i + b_i) ----
    sum6<<<blk(n64), 256, 0, stream>>>(c.T, c.ys[1], c.ys[2], c.ys[3], c.ys[4], c.ys[5], c.ys[6], n64);
    cvt_f2bf<<<blk(n64), 256, 0, stream>>>(c.T, c.bfA, n64);
    hipMemsetAsync(c.wpack, 0, (size_t)256 * 64 * 2, stream);   // 200 cols padded to 256
    for (int i = 0; i < 5; ++i) {
        pack_wt<<<blk((size_t)40 * 64), 256, 0, stream>>>(
            (const float*)d_in[20 + 2 * i], c.wpack + (size_t)i * 40 * 64, 64, 40, 40);
        copy_f32<<<1, 64, 0, stream>>>(c.bias200 + i * 40, (const float*)d_in[21 + 2 * i], 40);
    }
    {
        dim3 grid((n + 127) / 128, 4);
        wmma_gemm<<<grid, 256, 0, stream>>>(c.bfA, c.wpack, c.bias200, c.bigA, n, 64, 200, 1);
    }
    final_mix<<<n, 64, 0, stream>>>(c.bigA, c.wbuf + 5, (float*)d_out, n);
}
